// LinearAttention_31980326486722
// MI455X (gfx1250) — matrix operand order repaired and checked by reading, not compiled, not run on hardware
//
#include <hip/hip_runtime.h>
#include <hip/hip_bf16.h>
#include <cstdint>

#define HEADS   4
#define DHEAD   32
#define HIDDEN  128
#define CIN     128
#define F3      384
#define NPIX    16384      // 128*128 pixels per batch image
#define NBATCH  16
#define CHUNKS  8
#define QSCALE  0.17677669529663687f   // 32^-0.5

typedef __attribute__((ext_vector_type(16))) __bf16         v16bf;
typedef __attribute__((ext_vector_type(8)))  float          v8f;
typedef __attribute__((ext_vector_type(8)))  unsigned short us8;
typedef __attribute__((ext_vector_type(16))) unsigned short us16;

union FragB16 { v16bf v; us16 u; unsigned short s[16]; };

__device__ __forceinline__ unsigned short f2bf(float f) {
  unsigned int u = __float_as_uint(f);
  u += 0x7FFFu + ((u >> 16) & 1u);           // round-to-nearest-even
  return (unsigned short)(u >> 16);
}
__device__ __forceinline__ float bf2f(unsigned short h) {
  return __uint_as_float(((unsigned int)h) << 16);
}
__device__ __forceinline__ v8f wmma_bf16(const FragB16& a, const FragB16& b, v8f c) {
  return __builtin_amdgcn_wmma_f32_16x16x32_bf16(false, a.v, false, b.v, (short)0, c, false, false);
}

// ---------------------------------------------------------------------------
// Kernel 1: qkv = x @ w_qkv  (WMMA bf16, K=128), fused q feature-softmax.
//   q -> q_ws  [b][pix][h*32+d]      (bf16, row-major per pixel)
//   k -> k_ws  [(b*4+h)*32+d][pix]   (bf16, d-major: WMMA A-friendly)
//   v -> v_ws  [(b*4+h)*32+e][pix]   (bf16, e-major: WMMA B-friendly)
// ---------------------------------------------------------------------------
__global__ __launch_bounds__(256) void k_qkv(
    const float* __restrict__ x, const float* __restrict__ w_qkv,
    unsigned short* __restrict__ q_ws, unsigned short* __restrict__ k_ws,
    unsigned short* __restrict__ v_ws) {
  extern __shared__ unsigned short smem[];
  unsigned short* xs = smem;                 // [256][128] bf16 x tile
  unsigned short* wt = smem + 256 * CIN;     // [384][128] bf16 w_qkv^T

  const int t = threadIdx.x;
  const int b = blockIdx.x >> 6;
  const int pixbase = (blockIdx.x & 63) << 8;

  // stage x tile (one pixel row per thread, float4 loads -> bf16 LDS)
  {
    const float4* xr = (const float4*)(x + ((size_t)b * NPIX + pixbase + t) * CIN);
    unsigned short* xd = xs + t * CIN;
#pragma unroll
    for (int c4 = 0; c4 < CIN / 4; ++c4) {
      float4 v = xr[c4];
      xd[c4 * 4 + 0] = f2bf(v.x); xd[c4 * 4 + 1] = f2bf(v.y);
      xd[c4 * 4 + 2] = f2bf(v.z); xd[c4 * 4 + 3] = f2bf(v.w);
    }
  }
  // stage w_qkv transposed: wt[f][c]
  for (int i = t * 4; i < CIN * F3; i += 256 * 4) {
    int c = i / F3, f = i % F3;
    float4 v = *(const float4*)(w_qkv + i);
    wt[(f + 0) * CIN + c] = f2bf(v.x); wt[(f + 1) * CIN + c] = f2bf(v.y);
    wt[(f + 2) * CIN + c] = f2bf(v.z); wt[(f + 3) * CIN + c] = f2bf(v.w);
  }
  __syncthreads();

  const int wave = t >> 5, lane = t & 31;
  const int l16 = lane & 15, hi = (lane >> 4) & 1;
  const int mrow0 = wave * 32;               // 2 M-tiles (32 pixels) per wave

  for (int pair = 0; pair < 12; ++pair) {    // 12 N-pairs of 32 features
    const int fbase = pair * 32;
    v8f acc[2][2];
#pragma unroll
    for (int mt = 0; mt < 2; ++mt)
#pragma unroll
      for (int nt = 0; nt < 2; ++nt)
#pragma unroll
        for (int j = 0; j < 8; ++j) acc[mt][nt][j] = 0.0f;

#pragma unroll
    for (int ks = 0; ks < 4; ++ks) {         // K = 128 in 4 steps of 32
      FragB16 a[2], bb[2];
#pragma unroll
      for (int mt = 0; mt < 2; ++mt) {
        int arow = mrow0 + mt * 16 + l16;
        int ko = ks * 32 + hi * 16;
        *(us8*)&a[mt].s[0] = *(const us8*)&xs[arow * CIN + ko];
        *(us8*)&a[mt].s[8] = *(const us8*)&xs[arow * CIN + ko + 8];
      }
#pragma unroll
      for (int nt = 0; nt < 2; ++nt) {
        int col = fbase + nt * 16 + l16;
        bb[nt].u = *(const us16*)&wt[col * CIN + ks * 32 + hi * 16];
      }
#pragma unroll
      for (int mt = 0; mt < 2; ++mt)
#pragma unroll
        for (int nt = 0; nt < 2; ++nt)
          acc[mt][nt] = wmma_bf16(a[mt], bb[nt], acc[mt][nt]);
    }

    const int group = fbase >> 7;            // 0=q, 1=k, 2=v
    const int h = (fbase >> 5) & 3;

    if (group == 0) {
      // softmax over the 32 features of head h (across two tiles + 16 lanes)
#pragma unroll
      for (int mt = 0; mt < 2; ++mt) {
#pragma unroll
        for (int j = 0; j < 8; ++j) {
          float a0 = acc[mt][0][j], a1 = acc[mt][1][j];
          float m = fmaxf(a0, a1);
#pragma unroll
          for (int sft = 8; sft >= 1; sft >>= 1) m = fmaxf(m, __shfl_xor(m, sft, 16));
          float e0 = __expf(a0 - m), e1 = __expf(a1 - m);
          float s = e0 + e1;
#pragma unroll
          for (int sft = 8; sft >= 1; sft >>= 1) s += __shfl_xor(s, sft, 16);
          float inv = QSCALE / s;
          int pix = pixbase + mrow0 + mt * 16 + j + hi * 8;
          unsigned short* qp = q_ws + ((size_t)b * NPIX + pix) * HIDDEN + h * DHEAD + l16;
          qp[0]  = f2bf(e0 * inv);
          qp[16] = f2bf(e1 * inv);
        }
      }
    } else {
      unsigned short* dst = (group == 1) ? k_ws : v_ws;
      const size_t rb = (size_t)(b * HEADS + h) * DHEAD;
#pragma unroll
      for (int mt = 0; mt < 2; ++mt) {
#pragma unroll
        for (int j = 0; j < 8; ++j) {
          int pix = pixbase + mrow0 + mt * 16 + j + hi * 8;
          dst[(rb + l16)      * (size_t)NPIX + pix] = f2bf(acc[mt][0][j]);
          dst[(rb + 16 + l16) * (size_t)NPIX + pix] = f2bf(acc[mt][1][j]);
        }
      }
    }
  }
}

// ---------------------------------------------------------------------------
// Kernel 2: per (b,h,d) row of k: max and sum(exp) over n=16384
// ---------------------------------------------------------------------------
__global__ __launch_bounds__(256) void k_stats(const unsigned short* __restrict__ k_ws,
                                               float* __restrict__ m_ws,
                                               float* __restrict__ z_ws) {
  __shared__ float red[256];
  const int t = threadIdx.x;
  const unsigned short* row = k_ws + (size_t)blockIdx.x * NPIX;
  float m = -3.4e38f;
  for (int i = t; i < NPIX; i += 256) m = fmaxf(m, bf2f(row[i]));
  red[t] = m; __syncthreads();
  for (int s = 128; s > 0; s >>= 1) { if (t < s) red[t] = fmaxf(red[t], red[t + s]); __syncthreads(); }
  m = red[0]; __syncthreads();
  float z = 0.0f;
  for (int i = t; i < NPIX; i += 256) z += __expf(bf2f(row[i]) - m);
  red[t] = z; __syncthreads();
  for (int s = 128; s > 0; s >>= 1) { if (t < s) red[t] += red[t + s]; __syncthreads(); }
  if (t == 0) { m_ws[blockIdx.x] = m; z_ws[blockIdx.x] = red[0]; }
}

// ---------------------------------------------------------------------------
// Kernel 3: context[d,e] = sum_n softmax(k)[d,n] * v[e,n]   (WMMA over n)
// exp((k-m)) * (1/Z) folded into A-fragment build; partials via f32 atomics.
// Streaming phase: prefetch one cacheline ahead on both k and v rows.
// ---------------------------------------------------------------------------
__global__ __launch_bounds__(256) void k_ctx(const unsigned short* __restrict__ k_ws,
                                             const unsigned short* __restrict__ v_ws,
                                             const float* __restrict__ m_ws,
                                             const float* __restrict__ z_ws,
                                             float* __restrict__ ctx_ws) {
  const int bh = blockIdx.x / CHUNKS;
  const int chunk = blockIdx.x % CHUNKS;
  const int t = threadIdx.x, wave = t >> 5, lane = t & 31;
  const int l16 = lane & 15, hi = (lane >> 4) & 1;
  const int n0 = chunk * (NPIX / CHUNKS) + wave * 256;

  float mm[2], iz[2];
#pragma unroll
  for (int dt = 0; dt < 2; ++dt) {
    int d = dt * 16 + l16;
    mm[dt] = m_ws[bh * 32 + d];
    iz[dt] = 1.0f / z_ws[bh * 32 + d];
  }

  v8f acc[2][2];
#pragma unroll
  for (int dt = 0; dt < 2; ++dt)
#pragma unroll
    for (int et = 0; et < 2; ++et)
#pragma unroll
      for (int j = 0; j < 8; ++j) acc[dt][et][j] = 0.0f;

  for (int ks = 0; ks < 8; ++ks) {           // 256 n per wave, K=32 steps
    const int nb = n0 + ks * 32;
    if (ks < 6) {                            // prefetch 2 K-steps (128B) ahead
#pragma unroll
      for (int dt = 0; dt < 2; ++dt) {
        int d = dt * 16 + l16;
        __builtin_prefetch(k_ws + ((size_t)bh * 32 + d) * NPIX + nb + 64, 0, 3);
        __builtin_prefetch(v_ws + ((size_t)bh * 32 + d) * NPIX + nb + 64, 0, 3);
      }
    }
    FragB16 a[2], bb[2];
#pragma unroll
    for (int dt = 0; dt < 2; ++dt) {
      int d = dt * 16 + l16;
      const unsigned short* kr = k_ws + ((size_t)bh * 32 + d) * NPIX + nb + hi * 16;
#pragma unroll
      for (int u = 0; u < 8; ++u) a[dt].s[u]     = f2bf(__expf(bf2f(kr[u])      - mm[dt]) * iz[dt]);
#pragma unroll
      for (int u = 0; u < 8; ++u) a[dt].s[8 + u] = f2bf(__expf(bf2f(kr[8 + u]) - mm[dt]) * iz[dt]);
    }
#pragma unroll
    for (int et = 0; et < 2; ++et) {
      int e = et * 16 + l16;
      bb[et].u = *(const us16*)(v_ws + ((size_t)bh * 32 + e) * NPIX + nb + hi * 16);
    }
#pragma unroll
    for (int dt = 0; dt < 2; ++dt)
#pragma unroll
      for (int et = 0; et < 2; ++et)
        acc[dt][et] = wmma_bf16(a[dt], bb[et], acc[dt][et]);
  }

#pragma unroll
  for (int dt = 0; dt < 2; ++dt)
#pragma unroll
    for (int et = 0; et < 2; ++et)
#pragma unroll
      for (int j = 0; j < 8; ++j) {
        int d = dt * 16 + j + hi * 8;
        int e = et * 16 + l16;
        atomicAdd(&ctx_ws[((size_t)bh * 32 + d) * 32 + e], acc[dt][et][j]);
      }
}

// ---------------------------------------------------------------------------
// Kernel 4: hidden = ctx^T @ q per head (K=32 WMMA) -> LDS -> @ w_out + b_out
// ---------------------------------------------------------------------------
__global__ __launch_bounds__(256) void k_out(const unsigned short* __restrict__ q_ws,
                                             const float* __restrict__ ctx_ws,
                                             const float* __restrict__ w_out,
                                             const float* __restrict__ b_out,
                                             float* __restrict__ out) {
  extern __shared__ unsigned short smem[];
  unsigned short* ctxT = smem;               // [4][e=32][d=32] bf16 (transposed)
  unsigned short* wT   = smem + 4 * 32 * 32; // [col=128][row=128] bf16 w_out^T
  unsigned short* hid  = wT + 128 * 128;     // [256 pix][128] bf16
  __shared__ float bo[128];

  const int t = threadIdx.x;
  const int b = blockIdx.x >> 6;
  const int pixbase = (blockIdx.x & 63) << 8;

  for (int i = t; i < 4096; i += 256) {      // ctx -> bf16, transposed [h][e][d]
    int h = i >> 10, rem = i & 1023, d = rem >> 5, e = rem & 31;
    ctxT[h * 1024 + e * 32 + d] = f2bf(ctx_ws[(size_t)b * 4096 + i]);
  }
  for (int i = t * 4; i < 128 * 128; i += 1024) {  // w_out^T in bf16
    int r = i >> 7, c0 = i & 127;
    float4 v = *(const float4*)(w_out + i);
    wT[(c0 + 0) * 128 + r] = f2bf(v.x); wT[(c0 + 1) * 128 + r] = f2bf(v.y);
    wT[(c0 + 2) * 128 + r] = f2bf(v.z); wT[(c0 + 3) * 128 + r] = f2bf(v.w);
  }
  if (t < 128) bo[t] = b_out[t];
  __syncthreads();

  const int wave = t >> 5, lane = t & 31, l16 = lane & 15, hi = (lane >> 4) & 1;
  const int mrow0 = wave * 32;

  // stage 1: hid[pix][h*32+e] = sum_d ctx[h][d][e] * q[pix][h*32+d]
  for (int h = 0; h < HEADS; ++h) {
    FragB16 a[2], bb[2];
#pragma unroll
    for (int mt = 0; mt < 2; ++mt) {
      int lrow = mrow0 + mt * 16 + l16;
      const unsigned short* qp =
          q_ws + ((size_t)b * NPIX + pixbase + lrow) * HIDDEN + h * DHEAD + hi * 16;
      *(us8*)&a[mt].s[0] = *(const us8*)qp;
      *(us8*)&a[mt].s[8] = *(const us8*)(qp + 8);
    }
#pragma unroll
    for (int et = 0; et < 2; ++et) {
      int e = et * 16 + l16;
      bb[et].u = *(const us16*)&ctxT[h * 1024 + e * 32 + hi * 16];
    }
#pragma unroll
    for (int mt = 0; mt < 2; ++mt)
#pragma unroll
      for (int et = 0; et < 2; ++et) {
        v8f c;
#pragma unroll
        for (int j = 0; j < 8; ++j) c[j] = 0.0f;
        c = wmma_bf16(a[mt], bb[et], c);
#pragma unroll
        for (int j = 0; j < 8; ++j) {
          int lp = mrow0 + mt * 16 + j + hi * 8;
          hid[lp * 128 + h * DHEAD + et * 16 + l16] = f2bf(c[j]);
        }
      }
  }
  __syncthreads();

  // stage 2: out = hid @ w_out + b_out  (A-fragments hoisted across N-tiles)
#pragma unroll
  for (int mt = 0; mt < 2; ++mt) {
    FragB16 a[4];
    const int lrow = mrow0 + mt * 16 + l16;
#pragma unroll
    for (int ks = 0; ks < 4; ++ks) {
      int ko = ks * 32 + hi * 16;
      *(us8*)&a[ks].s[0] = *(const us8*)&hid[lrow * 128 + ko];
      *(us8*)&a[ks].s[8] = *(const us8*)&hid[lrow * 128 + ko + 8];
    }
    for (int nt = 0; nt < 8; ++nt) {
      v8f c;
#pragma unroll
      for (int j = 0; j < 8; ++j) c[j] = 0.0f;
      const int col = nt * 16 + l16;
#pragma unroll
      for (int ks = 0; ks < 4; ++ks) {
        FragB16 bb;
        bb.u = *(const us16*)&wT[col * 128 + ks * 32 + hi * 16];
        c = wmma_bf16(a[ks], bb, c);
      }
      float bias = bo[col];
#pragma unroll
      for (int j = 0; j < 8; ++j) {
        int pix = pixbase + mrow0 + mt * 16 + j + hi * 8;
        out[((size_t)b * NPIX + pix) * HIDDEN + col] = c[j] + bias;
      }
    }
  }
}

// ---------------------------------------------------------------------------
extern "C" void kernel_launch(void* const* d_in, const int* in_sizes, int n_in,
                              void* d_out, int out_size, void* d_ws, size_t ws_size,
                              hipStream_t stream) {
  const float* x     = (const float*)d_in[0];
  const float* w_qkv = (const float*)d_in[1];
  const float* w_out = (const float*)d_in[2];
  const float* b_out = (const float*)d_in[3];
  float* out = (float*)d_out;

  const size_t QKV_ELEMS = (size_t)NBATCH * NPIX * HIDDEN;  // 33,554,432 bf16 each
  unsigned short* q_ws = (unsigned short*)d_ws;
  unsigned short* k_ws = q_ws + QKV_ELEMS;
  unsigned short* v_ws = k_ws + QKV_ELEMS;
  float* m_ws   = (float*)(v_ws + QKV_ELEMS);
  float* z_ws   = m_ws + NBATCH * HEADS * DHEAD;
  float* ctx_ws = z_ws + NBATCH * HEADS * DHEAD;

  hipMemsetAsync(ctx_ws, 0, (size_t)NBATCH * HEADS * DHEAD * DHEAD * sizeof(float), stream);

  k_qkv<<<dim3(NBATCH * 64), dim3(256),
          (256 * CIN + F3 * CIN) * sizeof(unsigned short), stream>>>(x, w_qkv, q_ws, k_ws, v_ws);
  k_stats<<<dim3(NBATCH * HEADS * DHEAD), dim3(256), 0, stream>>>(k_ws, m_ws, z_ws);
  k_ctx<<<dim3(NBATCH * HEADS * CHUNKS), dim3(256), 0, stream>>>(k_ws, v_ws, m_ws, z_ws, ctx_ws);
  k_out<<<dim3(NBATCH * 64), dim3(256),
          (4 * 32 * 32 + 128 * 128 + 256 * 128) * sizeof(unsigned short), stream>>>(
      q_ws, ctx_ws, w_out, b_out, out);
}
